// CovariantAttention_28398323761430
// MI455X (gfx1250) — compile-verified, matrix operand order checked by reading, not run on hardware
//
#include <hip/hip_runtime.h>
#include <hip/hip_bf16.h>
#include <math.h>

// Problem dims (fixed by reference)
#define BB 4
#define NN 2048
#define DD 1024
#define DLL 64
#define DKK 128
#define DCC 8

typedef __attribute__((ext_vector_type(16))) _Float16 v16h;
typedef __attribute__((ext_vector_type(8)))  _Float16 v8h;
typedef __attribute__((ext_vector_type(8)))  float    v8f;
typedef __attribute__((ext_vector_type(4)))  float    v4f;

// ---------------------------------------------------------------------------
// Kernel A: per-batch prep. Computes
//   q = W_Q x + W_Qz z + Gamma(z,z)                       [128]
//   p_r = basis_r^T q, basis_r = skew combo of b/e/o      [8,128]
//   U[0] = W_K^T q, U[1+r] = W_K^T p_r  (rows 9..15 = 0)  [16,1024] -> f16
//   lam, tau (conformal temperature), sigma = exp(log_sigma)
// One block per batch, 128 threads.
// ---------------------------------------------------------------------------
__global__ __launch_bounds__(128) void prep_kernel(
    const float* __restrict__ xq, const float* __restrict__ zq,
    const float* __restrict__ W_Q, const float* __restrict__ W_Qz,
    const float* __restrict__ W_G, const float* __restrict__ W_K,
    const float* __restrict__ bb, const float* __restrict__ be,
    const float* __restrict__ bo, const float* __restrict__ log_sigma,
    _Float16* __restrict__ U_h, float* __restrict__ cst)
{
    const int b   = blockIdx.x;
    const int tid = threadIdx.x;
    __shared__ float sq[DKK];
    __shared__ float sp[DCC * DKK];
    __shared__ float sz[DLL];

    if (tid < DLL) sz[tid] = zq[b * DLL + tid];
    __syncthreads();

    // q[tid]
    {
        float acc = 0.f;
        const float* xqb = xq + b * DD;
        const float* wq  = W_Q + tid * DD;
        for (int d = 0; d < DD; ++d) acc += wq[d] * xqb[d];
        const float* wqz = W_Qz + tid * DLL;
        for (int l = 0; l < DLL; ++l) acc += wqz[l] * sz[l];
        const float* G = W_G + (size_t)tid * DLL * DLL;
        float qa = 0.f;
        for (int i = 0; i < DLL; ++i) {
            const float zi = sz[i];
            const float* Gi = G + i * DLL;
            for (int j = 0; j < DLL; ++j) qa += Gi[j] * sz[j] * zi;
        }
        sq[tid] = acc + qa;
    }
    __syncthreads();

    // p[r][i] = sum_j basis[r][j][i] * q[j]
    for (int it = tid; it < DCC * DKK; it += 128) {
        const int r = it >> 7;
        const int i = it & (DKK - 1);
        const float* bbr = bb + (size_t)r * DKK * DKK;
        const float* ber = be + (size_t)r * DKK * DKK;
        const float* bor = bo + (size_t)r * DKK * DKK;
        float s = 0.f;
        for (int j = 0; j < DKK; ++j) {
            const float Bji = 1.0f * (bbr[j * DKK + i] - bbr[i * DKK + j])
                            + 0.5f * (ber[j * DKK + i] - ber[i * DKK + j])
                            + 0.3f * (bor[j * DKK + i] - bor[i * DKK + j]);
            s += Bji * sq[j];
        }
        sp[it] = s;
    }
    __syncthreads();

    if (tid == 0) {
        float rsq = 0.f;
        for (int l = 0; l < DLL; ++l) rsq += sz[l] * sz[l];
        rsq = fminf(rsq, 1.0f - 1e-6f);
        const float lam = 2.0f / (1.0f - rsq + 1e-6f);
        cst[b * 2 + 0] = lam;
        cst[b * 2 + 1] = sqrtf((float)DKK) / lam;  // tau
        if (b == 0) cst[2 * BB] = __expf(log_sigma[0]);
    }

    // U rows: lanes sweep d contiguously -> coalesced W_K reads per i-iter
    _Float16* Ub = U_h + (size_t)b * 16 * DD;
    for (int d = tid; d < DD; d += 128) {
        float u[9];
        #pragma unroll
        for (int j = 0; j < 9; ++j) u[j] = 0.f;
        for (int i = 0; i < DKK; ++i) {
            const float wk = W_K[(size_t)i * DD + d];
            u[0] += wk * sq[i];
            #pragma unroll
            for (int r = 0; r < DCC; ++r) u[1 + r] += wk * sp[r * DKK + i];
        }
        #pragma unroll
        for (int j = 0; j < 9; ++j) Ub[j * DD + d] = (_Float16)u[j];
        #pragma unroll
        for (int j = 9; j < 16; ++j) Ub[j * DD + d] = (_Float16)0.f;
    }
}

// ---------------------------------------------------------------------------
// Kernel B: t[b,n,j] = sum_d X[b,n,d] * U[b,j,d]  via V_WMMA_F32_16X16X32_F16.
// One wave per 16-row n-tile; 256-thread blocks carry 8 tiles.
// Fragment K-mapping (ISA 7.12.2, 16-bit A 16x32):
//   lanes 0-15 : elem e -> K = (e<8 ? e : e+8)
//   lanes 16-31: elem e -> K = (e<8 ? e : e+8) + 8
// so each lane loads two contiguous 8-element chunks at d0+8*hi, d0+16+8*hi.
// Prefetch is unconditional: speculative prefetch past the buffer is
// silently dropped by hardware (ISA 10.5), so no per-iteration guard.
// ---------------------------------------------------------------------------
__global__ __launch_bounds__(256) void gemm_t_kernel(
    const float* __restrict__ xk, const _Float16* __restrict__ U_h,
    float* __restrict__ t)
{
    const int lane  = threadIdx.x & 31;
    const int wave  = threadIdx.x >> 5;
    const int tileg = blockIdx.x * 8 + wave;        // 0 .. B*N/16-1
    const int b     = tileg >> 7;                   // N/16 = 128 tiles/batch
    const int n0    = (tileg & 127) << 4;
    const int m     = lane & 15;
    const int hi    = lane >> 4;

    const float*    Xrow = xk  + ((size_t)(b * NN + n0 + m)) * DD;
    const _Float16* Urow = U_h + ((size_t)(b * 16 + m)) * DD;   // j == m

    v8f acc = {};
    for (int d0 = 0; d0 < DD; d0 += 32) {
        __builtin_prefetch(Xrow + d0 + 128, 0, 0);
        const int o1 = d0 + 8 * hi;
        const int o2 = d0 + 16 + 8 * hi;
        const v4f x0 = *(const v4f*)(Xrow + o1);
        const v4f x1 = *(const v4f*)(Xrow + o1 + 4);
        const v4f x2 = *(const v4f*)(Xrow + o2);
        const v4f x3 = *(const v4f*)(Xrow + o2 + 4);
        v16h afrag = {};
        #pragma unroll
        for (int e = 0; e < 4; ++e) {
            afrag[e]      = (_Float16)x0[e];
            afrag[4 + e]  = (_Float16)x1[e];
            afrag[8 + e]  = (_Float16)x2[e];
            afrag[12 + e] = (_Float16)x3[e];
        }
        const v8h u0 = *(const v8h*)(Urow + o1);
        const v8h u1 = *(const v8h*)(Urow + o2);
        v16h bfrag = {};
        #pragma unroll
        for (int e = 0; e < 8; ++e) { bfrag[e] = u0[e]; bfrag[8 + e] = u1[e]; }
        acc = __builtin_amdgcn_wmma_f32_16x16x32_f16(
            /*neg_a=*/false, afrag, /*neg_b=*/false, bfrag,
            /*c_mod=*/(short)0, acc, /*reuse_a=*/false, /*reuse_b=*/false);
    }
    // D layout: VGPR r -> M = r + 8*hi, lane%16 -> N
    #pragma unroll
    for (int r = 0; r < 8; ++r)
        t[((size_t)(b * NN + n0 + r + 8 * hi)) * 16 + m] = acc[r];
}

// ---------------------------------------------------------------------------
// Kernel C: scores + softmax + area screening. One block per batch.
//   coeff_r = W_delta . (z_q - z_k),  raw = t0 + sum_r coeff_r t_{1+r}
//   score = raw / tau ; screen = exp(-sigma * 0.5 lam^2 |delta|^2)
//   w = softmax(score) * screen
// ---------------------------------------------------------------------------
__global__ __launch_bounds__(256) void score_kernel(
    const float* __restrict__ zq, const float* __restrict__ zk,
    const float* __restrict__ W_delta, const float* __restrict__ t,
    const float* __restrict__ cst, float* __restrict__ w)
{
    const int b   = blockIdx.x;
    const int tid = threadIdx.x;
    __shared__ float sc[NN];
    __shared__ float scr[NN];
    __shared__ float red[256];
    __shared__ float szq[DLL];
    __shared__ float swd[DCC * DLL];

    if (tid < DLL) szq[tid] = zq[b * DLL + tid];
    for (int i = tid; i < DCC * DLL; i += 256) swd[i] = W_delta[i];
    __syncthreads();

    const float lam = cst[b * 2 + 0];
    const float tau = cst[b * 2 + 1];
    const float sig = cst[2 * BB];

    for (int n = tid; n < NN; n += 256) {
        const float* zkn = zk + ((size_t)(b * NN + n)) * DLL;
        float coeff[DCC];
        #pragma unroll
        for (int r = 0; r < DCC; ++r) coeff[r] = 0.f;
        float dsq = 0.f;
        for (int l = 0; l < DLL; ++l) {
            const float dl = szq[l] - zkn[l];
            dsq += dl * dl;
            #pragma unroll
            for (int r = 0; r < DCC; ++r) coeff[r] += swd[r * DLL + l] * dl;
        }
        const float* tn = t + ((size_t)(b * NN + n)) * 16;
        float raw = tn[0];
        #pragma unroll
        for (int r = 0; r < DCC; ++r) raw += coeff[r] * tn[1 + r];
        sc[n]  = raw / tau;
        scr[n] = __expf(-sig * 0.5f * lam * lam * dsq);
    }
    __syncthreads();

    float mx = -1e30f;
    for (int n = tid; n < NN; n += 256) mx = fmaxf(mx, sc[n]);
    red[tid] = mx; __syncthreads();
    for (int s = 128; s > 0; s >>= 1) {
        if (tid < s) red[tid] = fmaxf(red[tid], red[tid + s]);
        __syncthreads();
    }
    mx = red[0];
    __syncthreads();

    float sum = 0.f;
    for (int n = tid; n < NN; n += 256) {
        const float e = __expf(sc[n] - mx);
        sc[n] = e; sum += e;
    }
    red[tid] = sum; __syncthreads();
    for (int s = 128; s > 0; s >>= 1) {
        if (tid < s) red[tid] += red[tid + s];
        __syncthreads();
    }
    const float inv = 1.0f / red[0];
    for (int n = tid; n < NN; n += 256)
        w[b * NN + n] = sc[n] * inv * scr[n];
}

// ---------------------------------------------------------------------------
// Kernel Z: zero xa (ws is poisoned by harness; must clear every call).
// ---------------------------------------------------------------------------
__global__ __launch_bounds__(256) void zero_xa_kernel(float* __restrict__ xa)
{
    const int i = blockIdx.x * 256 + threadIdx.x;
    if (i < BB * DD) xa[i] = 0.f;
}

// ---------------------------------------------------------------------------
// Kernel D: xa[b,d] += sum_{n in chunk} w[b,n] * X[b,n,d]
// Split-N for memory-level parallelism: BB*32 blocks x 256 threads,
// each thread owns a float4 column slice, 64-row chunk, atomic finish.
// ---------------------------------------------------------------------------
__global__ __launch_bounds__(256) void wsum_kernel(
    const float* __restrict__ xk, const float* __restrict__ w,
    float* __restrict__ xa)
{
    const int b     = blockIdx.x >> 5;              // 32 chunks per batch
    const int chunk = blockIdx.x & 31;
    const int n0    = chunk * (NN / 32);            // 64 rows per chunk
    const int d4    = threadIdx.x;                  // float4 index, d = 4*d4

    const float* X  = xk + ((size_t)(b * NN + n0)) * DD + 4 * d4;
    const float* wb = w + b * NN + n0;

    v4f acc = {};
    for (int n = 0; n < NN / 32; ++n) {
        const float wn = wb[n];
        const v4f x = *(const v4f*)(X + (size_t)n * DD);
        acc += x * wn;
    }
    float* dst = xa + b * DD + 4 * d4;
    atomicAdd(dst + 0, acc[0]);
    atomicAdd(dst + 1, acc[1]);
    atomicAdd(dst + 2, acc[2]);
    atomicAdd(dst + 3, acc[3]);
}

// ---------------------------------------------------------------------------
// Kernel E: out[b] = W_O @ (W_V @ xa[b]).  One block per batch, 128 threads.
// ---------------------------------------------------------------------------
__global__ __launch_bounds__(128) void out_kernel(
    const float* __restrict__ W_V, const float* __restrict__ W_O,
    const float* __restrict__ xa, float* __restrict__ out)
{
    const int b   = blockIdx.x;
    const int tid = threadIdx.x;
    __shared__ float so[DKK];
    __shared__ float sxa[DD];
    for (int d = tid; d < DD; d += 128) sxa[d] = xa[b * DD + d];
    __syncthreads();
    {
        float acc = 0.f;
        const float* wv = W_V + (size_t)tid * DD;
        for (int d = 0; d < DD; ++d) acc += wv[d] * sxa[d];
        so[tid] = acc;
    }
    __syncthreads();
    for (int d = tid; d < DD; d += 128) {
        float s = 0.f;
        const float* wo = W_O + (size_t)d * DKK;
        #pragma unroll 8
        for (int k = 0; k < DKK; ++k) s += wo[k] * so[k];
        out[b * DD + d] = s;
    }
}

// ---------------------------------------------------------------------------
extern "C" void kernel_launch(void* const* d_in, const int* in_sizes, int n_in,
                              void* d_out, int out_size, void* d_ws, size_t ws_size,
                              hipStream_t stream) {
    const float* x_query   = (const float*)d_in[0];
    const float* z_query   = (const float*)d_in[1];
    const float* x_keys    = (const float*)d_in[2];
    const float* z_keys    = (const float*)d_in[3];
    const float* W_Q       = (const float*)d_in[4];
    const float* W_Qz      = (const float*)d_in[5];
    const float* W_Qgamma  = (const float*)d_in[6];
    const float* W_K       = (const float*)d_in[7];
    const float* W_V       = (const float*)d_in[8];
    const float* W_O       = (const float*)d_in[9];
    const float* W_delta   = (const float*)d_in[10];
    const float* basis_b   = (const float*)d_in[11];
    const float* basis_e   = (const float*)d_in[12];
    const float* basis_o   = (const float*)d_in[13];
    const float* log_sigma = (const float*)d_in[14];
    float* out = (float*)d_out;

    // Workspace layout (bytes)
    char* ws = (char*)d_ws;
    _Float16* U_h = (_Float16*)(ws);                 // B*16*1024 f16  = 131072 B
    float*    t   = (float*)(ws + 131072);           // B*N*16 f32    = 524288 B
    float*    w   = (float*)(ws + 655360);           // B*N f32       =  32768 B
    float*    xa  = (float*)(ws + 688128);           // B*1024 f32    =  16384 B
    float*    cst = (float*)(ws + 704512);           // {lam,tau}xB + sigma

    prep_kernel<<<BB, 128, 0, stream>>>(x_query, z_query, W_Q, W_Qz, W_Qgamma,
                                        W_K, basis_b, basis_e, basis_o,
                                        log_sigma, U_h, cst);
    gemm_t_kernel<<<(BB * (NN / 16)) / 8, 256, 0, stream>>>(x_keys, U_h, t);
    score_kernel<<<BB, 256, 0, stream>>>(z_query, z_keys, W_delta, t, cst, w);
    zero_xa_kernel<<<(BB * DD + 255) / 256, 256, 0, stream>>>(xa);
    wsum_kernel<<<BB * 32, 256, 0, stream>>>(x_keys, w, xa);
    out_kernel<<<BB, 128, 0, stream>>>(W_V, W_O, xa, out);
}
